// SS3DConv_45243185496332
// MI455X (gfx1250) — matrix operand order repaired and checked by reading, not compiled, not run on hardware
//
#include <hip/hip_runtime.h>
#include <hip/hip_bf16.h>
#include <stdint.h>

// ---------------------------------------------------------------------------
// Problem constants
// ---------------------------------------------------------------------------
#define BATCH      2
#define DMODEL     128
#define DINNER     256
#define CCONV      16
#define NEIGH      27
#define DSEQ       432
#define DSEQ_P     448        // DSEQ padded to multiple of 32
#define DSTATE     16
#define DTRANK     14
#define DTRANK_P   32         // padded K for the dt GEMM
#define LSPAT      4096
#define MROWS      8192
#define XDBL_LD    48         // 46 padded to 48 (f32 buffer)

typedef __attribute__((ext_vector_type(16))) __bf16 bf16x16;
typedef __attribute__((ext_vector_type(8)))  float  v8f;
typedef __attribute__((ext_vector_type(4)))  unsigned int u32x4;
typedef __attribute__((ext_vector_type(8)))  int    i32x8;
typedef __attribute__((ext_vector_type(4)))  int    i32x4;

static __device__ __forceinline__ __bf16 f32_to_bf16(float f) {
    union { float f; unsigned u; } x; x.f = f;
    unsigned r = (x.u + 0x7FFFu + ((x.u >> 16) & 1u)) >> 16;   // RNE
    union { unsigned short s; __bf16 b; } y; y.s = (unsigned short)r;
    return y.b;
}

static __device__ __forceinline__ float silu(float v) {
    return v / (1.0f + __expf(-v));
}

// ---------------------------------------------------------------------------
// TDM: issue a 2-D tensor-tile load into LDS (6-arg clang-23 builtin form).
//   tensor is (dim1 rows x dim0 cols) of bf16, row stride = stride0 elements.
//   tile is (tile1 rows x tile0 cols), landing contiguously at lds_addr.
// D# layout per CDNA5 ISA 8.3/8.4.
// ---------------------------------------------------------------------------
static __device__ __forceinline__ void tdm_load_2d(
    unsigned lds_addr, unsigned long long gaddr,
    unsigned dim0, unsigned dim1,
    unsigned tile0, unsigned tile1,
    unsigned long long stride0)
{
    u32x4 g0;
    g0.x = 1u;                                              // count=1, user D#
    g0.y = lds_addr;                                        // LDS byte address
    g0.z = (unsigned)(gaddr & 0xFFFFFFFFull);               // global_addr lo
    g0.w = (unsigned)((gaddr >> 32) & 0x01FFFFFFull)        // global_addr hi
         | (2u << 30);                                      // type = 2 (image)
    i32x8 g1;
    g1.s0 = (int)(1u << 16);                                // data_size = 2 B
    g1.s1 = (int)((dim0 & 0xFFFFu) << 16);                  // tensor_dim0 lo16
    g1.s2 = (int)((dim0 >> 16) | ((dim1 & 0xFFFFu) << 16)); // dim0 hi / dim1 lo
    g1.s3 = (int)((dim1 >> 16) | (tile0 << 16));            // dim1 hi / tile0
    g1.s4 = (int)(tile1);                                   // tile1 (tile2=0)
    g1.s5 = (int)(unsigned)(stride0 & 0xFFFFFFFFull);       // dim0_stride lo32
    g1.s6 = (int)(unsigned)((stride0 >> 32) & 0xFFFFull);   // stride hi (d1s=0)
    g1.s7 = 0;
    i32x4 g2 = (i32x4){0, 0, 0, 0};
    i32x4 g3 = (i32x4){0, 0, 0, 0};
    i32x8 g4 = (i32x8){0, 0, 0, 0, 0, 0, 0, 0};
    __builtin_amdgcn_tensor_load_to_lds(g0, g1, g2, g3, g4, 0);
}

// ---------------------------------------------------------------------------
// WMMA GEMM, double-buffered TDM staging.
//   C[M x N] (f32) = A[M x Kpad] (bf16, zero-padded) x Bt[Kpad x Npad] (bf16)
// Block tile 128x128, 8 waves, each wave owns 32x64 (2x4 WMMA frags).
// Wave 0 issues TDM loads for tile k+1, then s_wait_tensorcnt 2 so the DMA of
// the next tile overlaps the WMMAs on the current tile; barrier releases the
// workgroup. Operands are pre-padded so staging needs no guards.
// ---------------------------------------------------------------------------
#define TILE_M 128
#define TILE_N 128
#define TILE_K 32

__global__ __launch_bounds__(256)
void gemm_bf16_wmma_tdm(const __bf16* __restrict__ A, int Kpad,
                        const __bf16* __restrict__ Bt, int Npad,
                        float* __restrict__ C, int ldc,
                        int M, int N)
{
    __shared__ __align__(16) __bf16 lA[2][TILE_M * TILE_K];   // 2 x 8 KB
    __shared__ __align__(16) __bf16 lB[2][TILE_K * TILE_N];   // 2 x 8 KB

    const int tid  = threadIdx.x;
    const int lane = tid & 31;
    const int wave = tid >> 5;
    const int wm   = wave & 3;       // 4 waves along M
    const int wn   = wave >> 2;      // 2 waves along N

    const int blockM = blockIdx.y * TILE_M;
    const int blockN = blockIdx.x * TILE_N;

    v8f acc[2][4];
    #pragma unroll
    for (int i = 0; i < 2; ++i)
        #pragma unroll
        for (int j = 0; j < 4; ++j) acc[i][j] = (v8f){};

    const unsigned ldsA0 = (unsigned)(uintptr_t)(&lA[0][0]);
    const unsigned ldsA1 = (unsigned)(uintptr_t)(&lA[1][0]);
    const unsigned ldsB0 = (unsigned)(uintptr_t)(&lB[0][0]);
    const unsigned ldsB1 = (unsigned)(uintptr_t)(&lB[1][0]);
    const unsigned long long gA = (unsigned long long)(uintptr_t)A;
    const unsigned long long gB = (unsigned long long)(uintptr_t)Bt;

    const int m16  = lane & 15;
    const int half = lane >> 4;
    const int nk   = Kpad / TILE_K;

    // prologue: DMA tile 0 into buffer 0
    if (wave == 0) {
        tdm_load_2d(ldsA0, gA + 2ull * ((unsigned long long)blockM * Kpad),
                    (unsigned)Kpad, (unsigned)M, TILE_K, TILE_M,
                    (unsigned long long)Kpad);
        tdm_load_2d(ldsB0, gB + 2ull * (unsigned long long)blockN,
                    (unsigned)Npad, (unsigned)Kpad, TILE_N, TILE_K,
                    (unsigned long long)Npad);
    }

    for (int kt = 0; kt < nk; ++kt) {
        const int cur = kt & 1;
        if (wave == 0) {
            if (kt + 1 < nk) {
                // DMA next tile into the other buffer, then wait until only
                // those 2 newest DMAs remain in flight (current tile landed).
                const int k1 = (kt + 1) * TILE_K;
                tdm_load_2d(cur ? ldsA0 : ldsA1,
                            gA + 2ull * ((unsigned long long)blockM * Kpad + k1),
                            (unsigned)Kpad, (unsigned)M, TILE_K, TILE_M,
                            (unsigned long long)Kpad);
                tdm_load_2d(cur ? ldsB0 : ldsB1,
                            gB + 2ull * ((unsigned long long)k1 * Npad + blockN),
                            (unsigned)Npad, (unsigned)Kpad, TILE_N, TILE_K,
                            (unsigned long long)Npad);
                __builtin_amdgcn_s_wait_tensorcnt(2);
            } else {
                __builtin_amdgcn_s_wait_tensorcnt(0);
            }
        }
        __syncthreads();

        // A frag: lane -> row m; elems 0..7 -> K=half*8+i, 8..15 -> 16+half*8+i
        bf16x16 afrag[2];
        #pragma unroll
        for (int fm = 0; fm < 2; ++fm) {
            const __bf16* p = &lA[cur][(wm * 32 + fm * 16 + m16) * TILE_K];
            #pragma unroll
            for (int i = 0; i < 8; ++i) {
                afrag[fm][i]     = p[half * 8 + i];
                afrag[fm][8 + i] = p[16 + half * 8 + i];
            }
        }
        // B frag: lane -> col n; elems 0..7 -> K=half*8+i, 8..15 -> 16+half*8+i
        bf16x16 bfrag[4];
        #pragma unroll
        for (int fn = 0; fn < 4; ++fn) {
            const __bf16* p = &lB[cur][wn * 64 + fn * 16 + m16];
            #pragma unroll
            for (int i = 0; i < 8; ++i) {
                bfrag[fn][i]     = p[(half * 8 + i) * TILE_N];
                bfrag[fn][8 + i] = p[(16 + half * 8 + i) * TILE_N];
            }
        }

        #pragma unroll
        for (int fm = 0; fm < 2; ++fm)
            #pragma unroll
            for (int fn = 0; fn < 4; ++fn)
                acc[fm][fn] = __builtin_amdgcn_wmma_f32_16x16x32_bf16(
                    false, afrag[fm], false, bfrag[fn],
                    (short)0, acc[fm][fn], false, false);

        __syncthreads();   // all waves done reading buf[cur] before overwrite
    }

    // C store: lanes 0-15 -> N=lane, M=r ; lanes 16-31 -> M=8+r
    const int cn  = lane & 15;
    const int mhi = (lane >> 4) * 8;
    const int rowb = blockM + wm * 32;
    const int colb = blockN + wn * 64;
    if ((blockM + TILE_M) <= M && (blockN + TILE_N) <= N) {
        // full tile: unguarded stores
        #pragma unroll
        for (int fm = 0; fm < 2; ++fm)
            #pragma unroll
            for (int fn = 0; fn < 4; ++fn)
                #pragma unroll
                for (int r = 0; r < 8; ++r)
                    C[(size_t)(rowb + fm * 16 + mhi + r) * ldc
                      + (colb + fn * 16 + cn)] = acc[fm][fn][r];
    } else {
        #pragma unroll
        for (int fm = 0; fm < 2; ++fm)
            #pragma unroll
            for (int fn = 0; fn < 4; ++fn)
                #pragma unroll
                for (int r = 0; r < 8; ++r) {
                    const int grow = rowb + fm * 16 + mhi + r;
                    const int gcol = colb + fn * 16 + cn;
                    if (grow < M && gcol < N)
                        C[(size_t)grow * ldc + gcol] = acc[fm][fn][r];
                }
    }
}

// ---------------------------------------------------------------------------
// f32 -> bf16 flat convert (for the x input)
// ---------------------------------------------------------------------------
__global__ void cvt_f32_bf16_kernel(const float* __restrict__ in,
                                    __bf16* __restrict__ out, int n)
{
    const int idx = blockIdx.x * blockDim.x + threadIdx.x;
    if (idx < n) out[idx] = f32_to_bf16(in[idx]);
}

// ---------------------------------------------------------------------------
// Weight (N x K f32) -> transposed padded bf16 (Kpad x Npad), zero-fill pad.
// ---------------------------------------------------------------------------
__global__ void wt_transpose_kernel(const float* __restrict__ in, int N, int K,
                                    __bf16* __restrict__ out, int Kpad, int Npad)
{
    const int idx = blockIdx.x * blockDim.x + threadIdx.x;
    if (idx >= Kpad * Npad) return;
    const int k = idx / Npad, n = idx % Npad;
    const float v = (k < K && n < N) ? in[(size_t)n * K + k] : 0.0f;
    out[idx] = f32_to_bf16(v);
}

// ---------------------------------------------------------------------------
// Extract dt-rank columns of xdbl (f32, ld 48) -> bf16 (MROWS x 32, zero pad)
// ---------------------------------------------------------------------------
__global__ void dtr_extract_kernel(const float* __restrict__ xdbl,
                                   __bf16* __restrict__ dtr)
{
    const int idx = blockIdx.x * blockDim.x + threadIdx.x;
    if (idx >= MROWS * DTRANK_P) return;
    const int row = idx >> 5, kk = idx & 31;
    const float v = (kk < DTRANK) ? xdbl[(size_t)row * XDBL_LD + kk] : 0.0f;
    dtr[idx] = f32_to_bf16(v);
}

// ---------------------------------------------------------------------------
// Zero the pad columns (DSEQ..DSEQ_P) of a (MROWS x DSEQ_P) bf16 buffer.
// ---------------------------------------------------------------------------
__global__ void pad_zero_kernel(__bf16* __restrict__ buf)
{
    const int idx = blockIdx.x * blockDim.x + threadIdx.x;
    if (idx >= MROWS * (DSEQ_P - DSEQ)) return;
    const int row = idx / (DSEQ_P - DSEQ);
    const int col = DSEQ + idx % (DSEQ_P - DSEQ);
    buf[(size_t)row * DSEQ_P + col] = f32_to_bf16(0.0f);
}

// ---------------------------------------------------------------------------
// Grouped 3x3x3 conv (256 -> 16 ch, 16 groups) + bias + SiLU.  f32 in/out.
// ---------------------------------------------------------------------------
__global__ void conv3d_silu_kernel(const float* __restrict__ xz,
                                   const float* __restrict__ cw,
                                   const float* __restrict__ cb,
                                   float* __restrict__ xc)
{
    const int idx = blockIdx.x * blockDim.x + threadIdx.x;
    if (idx >= BATCH * CCONV * LSPAT) return;
    const int w = idx & 15, h = (idx >> 4) & 15, d = (idx >> 8) & 15;
    const int c = (idx >> 12) & 15, b = idx >> 16;

    float sum = cb[c];
    for (int kd = 0; kd < 3; ++kd) {
        const int dz = d + kd - 1; if ((unsigned)dz >= 16u) continue;
        for (int kh = 0; kh < 3; ++kh) {
            const int hz = h + kh - 1; if ((unsigned)hz >= 16u) continue;
            for (int kw = 0; kw < 3; ++kw) {
                const int wz = w + kw - 1; if ((unsigned)wz >= 16u) continue;
                const size_t row = (size_t)b * LSPAT + ((dz * 16 + hz) * 16 + wz);
                const float* xp = xz + row * 512 + c * 16;
                const int tap = (kd * 3 + kh) * 3 + kw;
                #pragma unroll
                for (int i = 0; i < 16; ++i)
                    sum += xp[i] * cw[(c * 16 + i) * 27 + tap];
            }
        }
    }
    xc[idx] = silu(sum);
}

// ---------------------------------------------------------------------------
// Unfold 27-neighborhood; writes bf16 rows of length DSEQ_P (pad zeroed).
// ---------------------------------------------------------------------------
__global__ void unfold3_kernel(const float* __restrict__ xc,
                               __bf16* __restrict__ xsT)
{
    const int row = blockIdx.x * blockDim.x + threadIdx.x;
    if (row >= MROWS) return;
    const int l = row & (LSPAT - 1), b = row >> 12;
    const int w = l & 15, h = (l >> 4) & 15, d = (l >> 8) & 15;
    __bf16* dst = xsT + (size_t)row * DSEQ_P;
    for (int c = 0; c < CCONV; ++c) {
        const float* src = xc + (size_t)(b * CCONV + c) * LSPAT;
        int t = 0;
        for (int i = 0; i < 3; ++i) {
            const int dz = d + i - 1;
            for (int j = 0; j < 3; ++j) {
                const int hz = h + j - 1;
                for (int k = 0; k < 3; ++k, ++t) {
                    const int wz = w + k - 1;
                    float v = 0.0f;
                    if ((unsigned)dz < 16u && (unsigned)hz < 16u && (unsigned)wz < 16u)
                        v = src[(dz * 16 + hz) * 16 + wz];
                    dst[c * 27 + t] = f32_to_bf16(v);
                }
            }
        }
    }
    #pragma unroll
    for (int p = DSEQ; p < DSEQ_P; ++p) dst[p] = f32_to_bf16(0.0f);
}

// ---------------------------------------------------------------------------
// Selective scan; one thread per (b, d, n); shfl_xor(width 16) state reduce.
// ---------------------------------------------------------------------------
__global__ void scan_kernel(const float* __restrict__ dts,    // MROWS x DSEQ
                            const float* __restrict__ xdbl,   // MROWS x XDBL_LD
                            const __bf16* __restrict__ xsT,   // MROWS x DSEQ_P
                            const float* __restrict__ A_logs, // DSEQ x DSTATE
                            const float* __restrict__ dt_b,
                            const float* __restrict__ Ds,
                            __bf16* __restrict__ yT)          // MROWS x DSEQ_P
{
    const int tid = blockIdx.x * blockDim.x + threadIdx.x;
    if (tid >= BATCH * DSEQ * DSTATE) return;
    const int n   = tid & 15;
    const int seq = tid >> 4;
    const int d   = seq % DSEQ;
    const int b   = seq / DSEQ;

    const float An  = -__expf(A_logs[d * DSTATE + n]);
    const float dtb = dt_b[d];
    const float Dsk = Ds[d];

    float h = 0.0f;
    for (int l = 0; l < LSPAT; ++l) {
        const size_t row = (size_t)b * LSPAT + l;
        const float draw  = dts[row * DSEQ + d] + dtb;
        const float delta = (draw > 20.0f) ? draw : __logf(1.0f + __expf(draw));
        const float u  = (float)xsT[row * DSEQ_P + d];
        const float Bv = xdbl[row * XDBL_LD + DTRANK + n];
        const float Cv = xdbl[row * XDBL_LD + DTRANK + DSTATE + n];
        const float a  = __expf(delta * An);
        h = a * h + (delta * u) * Bv;
        float val = h * Cv;
        val += __shfl_xor(val, 8, 16);
        val += __shfl_xor(val, 4, 16);
        val += __shfl_xor(val, 2, 16);
        val += __shfl_xor(val, 1, 16);
        if (n == 0) yT[row * DSEQ_P + d] = f32_to_bf16(val + Dsk * u);
    }
}

// ---------------------------------------------------------------------------
// LayerNorm(256) + SiLU(z) gate -> bf16 g.  One 256-thread block per row.
// ---------------------------------------------------------------------------
__global__ __launch_bounds__(256)
void ln_gate_kernel(const float* __restrict__ yf,   // MROWS x DINNER
                    const float* __restrict__ xz,   // MROWS x 512
                    const float* __restrict__ ln_g,
                    const float* __restrict__ ln_b,
                    __bf16* __restrict__ g)
{
    const int row = blockIdx.x;
    const int t   = threadIdx.x;
    const float v = yf[(size_t)row * DINNER + t];

    float s = v, s2 = v * v;
    #pragma unroll
    for (int off = 16; off >= 1; off >>= 1) {
        s  += __shfl_xor(s,  off, 32);
        s2 += __shfl_xor(s2, off, 32);
    }
    __shared__ float ws1[8], ws2[8];
    __shared__ float mu_s, rstd_s;
    const int lane = t & 31, wave = t >> 5;
    if (lane == 0) { ws1[wave] = s; ws2[wave] = s2; }
    __syncthreads();
    if (t == 0) {
        float S = 0.0f, S2 = 0.0f;
        #pragma unroll
        for (int i = 0; i < 8; ++i) { S += ws1[i]; S2 += ws2[i]; }
        const float mu  = S  * (1.0f / DINNER);
        const float var = S2 * (1.0f / DINNER) - mu * mu;
        mu_s = mu;
        rstd_s = rsqrtf(var + 1e-5f);
    }
    __syncthreads();
    const float ynorm = (v - mu_s) * rstd_s * ln_g[t] + ln_b[t];
    const float z = xz[(size_t)row * 512 + DINNER + t];
    g[(size_t)row * DINNER + t] = f32_to_bf16(ynorm * silu(z));
}

// ---------------------------------------------------------------------------
// Launch
// ---------------------------------------------------------------------------
extern "C" void kernel_launch(void* const* d_in, const int* in_sizes, int n_in,
                              void* d_out, int out_size, void* d_ws, size_t ws_size,
                              hipStream_t stream)
{
    const float* x        = (const float*)d_in[0];
    const float* w_in     = (const float*)d_in[1];
    const float* conv_w   = (const float*)d_in[2];
    const float* conv_b   = (const float*)d_in[3];
    const float* x_proj_w = (const float*)d_in[4];
    const float* dt_w     = (const float*)d_in[5];
    const float* dt_b     = (const float*)d_in[6];
    const float* A_logs   = (const float*)d_in[7];
    const float* Ds       = (const float*)d_in[8];
    const float* w_fold   = (const float*)d_in[9];
    const float* ln_g     = (const float*)d_in[10];
    const float* ln_b     = (const float*)d_in[11];
    const float* w_out    = (const float*)d_in[12];
    float* out = (float*)d_out;

    // ---- workspace layout ----
    char* cur = (char*)d_ws;
    auto alloc_f32 = [&](size_t n) { float*  p = (float*)cur;  cur += n * 4;              return p; };
    auto alloc_bf  = [&](size_t n) { __bf16* p = (__bf16*)cur; cur += ((n * 2 + 255) & ~255ull); return p; };

    float*  xz    = alloc_f32((size_t)MROWS * 512);
    float*  xc    = alloc_f32((size_t)BATCH * CCONV * LSPAT);
    float*  xdbl  = alloc_f32((size_t)MROWS * XDBL_LD);
    float*  dts   = alloc_f32((size_t)MROWS * DSEQ);
    float*  yf    = alloc_f32((size_t)MROWS * DINNER);
    __bf16* x_bf  = alloc_bf((size_t)MROWS * DMODEL);
    __bf16* xsT   = alloc_bf((size_t)MROWS * DSEQ_P);
    __bf16* dtr   = alloc_bf((size_t)MROWS * DTRANK_P);
    __bf16* yT    = alloc_bf((size_t)MROWS * DSEQ_P);
    __bf16* gbuf  = alloc_bf((size_t)MROWS * DINNER);
    __bf16* w_inT   = alloc_bf((size_t)128 * 512);
    __bf16* xprojT  = alloc_bf((size_t)DSEQ_P * 128);
    __bf16* dtwT    = alloc_bf((size_t)DTRANK_P * 512);
    __bf16* wfoldT  = alloc_bf((size_t)DSEQ_P * 256);
    __bf16* woutT   = alloc_bf((size_t)256 * 128);

    const dim3 blk(256);

    // ---- operand preparation (bf16, padded) ----
    cvt_f32_bf16_kernel<<<dim3((MROWS * DMODEL) / 256), blk, 0, stream>>>(
        x, x_bf, MROWS * DMODEL);
    wt_transpose_kernel<<<dim3((128 * 512) / 256), blk, 0, stream>>>(
        w_in, 512, DMODEL, w_inT, 128, 512);
    wt_transpose_kernel<<<dim3((DSEQ_P * 128) / 256), blk, 0, stream>>>(
        x_proj_w, DTRANK + 2 * DSTATE, DSEQ, xprojT, DSEQ_P, 128);
    wt_transpose_kernel<<<dim3((DTRANK_P * 512) / 256), blk, 0, stream>>>(
        dt_w, DSEQ, DTRANK, dtwT, DTRANK_P, 512);
    wt_transpose_kernel<<<dim3((DSEQ_P * 256) / 256), blk, 0, stream>>>(
        w_fold, DINNER, DSEQ, wfoldT, DSEQ_P, 256);
    wt_transpose_kernel<<<dim3((256 * 128) / 256), blk, 0, stream>>>(
        w_out, DMODEL, DINNER, woutT, 256, 128);

    // 1) xz = x @ w_in^T          (8192 x 128 x 512)
    gemm_bf16_wmma_tdm<<<dim3(4, 64), blk, 0, stream>>>(
        x_bf, DMODEL, w_inT, 512, xz, 512, MROWS, 512);

    // 2) grouped conv3d + SiLU
    conv3d_silu_kernel<<<dim3((BATCH * CCONV * LSPAT) / 256), blk, 0, stream>>>(
        xz, conv_w, conv_b, xc);

    // 3) unfold -> xsT bf16 (8192 x 448, padded)
    unfold3_kernel<<<dim3(MROWS / 256), blk, 0, stream>>>(xc, xsT);

    // 4) x_dbl = xs @ x_proj_w^T  (8192 x 448 x 46->128)
    gemm_bf16_wmma_tdm<<<dim3(1, 64), blk, 0, stream>>>(
        xsT, DSEQ_P, xprojT, 128, xdbl, XDBL_LD, MROWS, DTRANK + 2 * DSTATE);

    // 5) dt-rank extract + dts = dtr @ dt_w^T  (8192 x 32 x 432->512)
    dtr_extract_kernel<<<dim3((MROWS * DTRANK_P) / 256), blk, 0, stream>>>(xdbl, dtr);
    gemm_bf16_wmma_tdm<<<dim3(4, 64), blk, 0, stream>>>(
        dtr, DTRANK_P, dtwT, 512, dts, DSEQ, MROWS, DSEQ);

    // 6) selective scan -> yT bf16 (pad cols zeroed first)
    pad_zero_kernel<<<dim3((MROWS * (DSEQ_P - DSEQ)) / 256), blk, 0, stream>>>(yT);
    scan_kernel<<<dim3((BATCH * DSEQ * DSTATE + 255) / 256), blk, 0, stream>>>(
        dts, xdbl, xsT, A_logs, dt_b, Ds, yT);

    // 7) yf = yT @ w_fold^T       (8192 x 448 x 256)
    gemm_bf16_wmma_tdm<<<dim3(2, 64), blk, 0, stream>>>(
        yT, DSEQ_P, wfoldT, 256, yf, DINNER, MROWS, DINNER);

    // 8) LayerNorm + SiLU gate -> g bf16
    ln_gate_kernel<<<dim3(MROWS), blk, 0, stream>>>(yf, xz, ln_g, ln_b, gbuf);

    // 9) out = g @ w_out^T        (8192 x 256 x 128)
    gemm_bf16_wmma_tdm<<<dim3(1, 64), blk, 0, stream>>>(
        gbuf, DINNER, woutT, 128, out, DMODEL, MROWS, DMODEL);
}
